// _LSTM_1194000908835
// MI455X (gfx1250) — compile-verified, matrix operand order checked by reading, not run on hardware
//
#include <hip/hip_runtime.h>
#include <stdint.h>

// MI455X / gfx1250 two-layer LSTM + projection.
//  - bf16 WMMA, f32 accumulate, f32 cell state.
//  - 32 persistent blocks (1 hidden-col tile each, 4 waves = 4 batch tiles).
//  - Recurrent weights resident in LDS (192KB) for all 512 steps; activations
//    TDM-staged into the remaining 128KB each step. 320KB dynamic LDS total.
//  - Grid barrier x2/step, ping-pong h buffers.
//  - Output projection: separate fully parallel WMMA kernel.

#define NB 64      // batch
#define NT 512     // seq len
#define ND 128     // input size
#define NH 512     // hidden
#define NG 2048    // 4*H
#define NO 1024    // D*P
#define NWG 32     // persistent blocks (one per hidden-col tile)
#define TPB 128    // 4 waves
#define LDS_BYTES 327680u   // 320 KB

typedef __attribute__((ext_vector_type(16))) __bf16 v16bf;
typedef __attribute__((ext_vector_type(8)))  float  v8f;

__device__ __forceinline__ unsigned short f2bf(float f) {
  union { float f; unsigned u; } c; c.f = f;
  unsigned r = c.u + 0x7FFFu + ((c.u >> 16) & 1u);  // RNE
  return (unsigned short)(r >> 16);
}
__device__ __forceinline__ float sigm(float x) { return 1.0f / (1.0f + __expf(-x)); }

// 16x32 bf16 WMMA operand fragment (A or B): lane row = lane&15, lanes 16-31
// take the +8 K-half of each 16-K block; two contiguous 16B loads per lane.
template <typename PT>
__device__ __forceinline__ v16bf ldfrag(PT p, int lane) {
  union { uint4 q[2]; v16bf v; } f;
  const int off = ((lane >> 4) & 1) * 8;
  f.q[0] = *(const uint4*)(p + off);
  f.q[1] = *(const uint4*)(p + off + 16);
  return f.v;
}

#define WMMA_BF16(A, Bm, C) \
  __builtin_amdgcn_wmma_f32_16x16x32_bf16(false, (A), false, (Bm), (short)0, (C), false, false)

// ---------------- Tensor Data Mover (async global->LDS DMA) ----------------
#if __has_builtin(__builtin_amdgcn_tensor_load_to_lds) && \
    __has_builtin(__builtin_amdgcn_s_wait_tensorcnt)
#define HAVE_TDM 1
typedef __attribute__((ext_vector_type(4))) unsigned tdm_u4;
typedef __attribute__((ext_vector_type(8))) int      tdm_i8;
typedef __attribute__((ext_vector_type(4))) int      tdm_i4;

// 2D tile load: tile1 rows of tile0 bf16 elements, row stride stride0 elems,
// packed contiguously into LDS at byte offset lds_off (dynamic LDS base = 0).
// D# layout per CDNA5 ISA §8: group0 = {count, lds_addr, global_addr, type=2},
// group1 = {data_size=2B, tensor dims, tile dims, dim0 stride}.
__device__ __forceinline__ void tdm_load_2d(unsigned lds_off, const void* gptr,
                                            unsigned tile0, unsigned tile1,
                                            unsigned long long stride0) {
  unsigned long long ga = (unsigned long long)(uintptr_t)gptr;
  tdm_u4 g0;
  g0.x = 1u;                                          // count = 1
  g0.y = lds_off;                                     // lds byte address
  g0.z = (unsigned)ga;                                // global addr [95:64]
  g0.w = (unsigned)((ga >> 32) & 0x01FFFFFFu) | (2u << 30);  // [120:96] | type=2
  unsigned td0 = (unsigned)stride0; if (td0 < tile0) td0 = tile0;  // dim0 extent
  unsigned td1 = tile1;
  tdm_i8 g1;
  g1[0] = (int)(1u << 16);                            // data_size = 1 (2 bytes)
  g1[1] = (int)((td0 & 0xFFFFu) << 16);               // tensor_dim0 [79:48] lo
  g1[2] = (int)((td0 >> 16) | ((td1 & 0xFFFFu) << 16));
  g1[3] = (int)((td1 >> 16) | ((tile0 & 0xFFFFu) << 16));  // tile_dim0 [127:112]
  g1[4] = (int)(tile1 & 0xFFFFu);                     // tile_dim1 [143:128]
  g1[5] = (int)(unsigned)(stride0 & 0xFFFFFFFFu);     // dim0_stride [207:160]
  g1[6] = (int)(unsigned)((stride0 >> 32) & 0xFFFFu);
  g1[7] = 0;
  tdm_i4 z4 = (tdm_i4)0;
#if defined(__clang_major__) && __clang_major__ >= 23
  tdm_i8 z8 = (tdm_i8)0;
  __builtin_amdgcn_tensor_load_to_lds(g0, g1, z4, z4, z8, 0);
#else
  __builtin_amdgcn_tensor_load_to_lds(g0, g1, z4, z4, 0);
#endif
}
#endif

// Monotonic grid barrier across NWG resident blocks.
__device__ __forceinline__ void gridbar(unsigned* bar, int tid) {
  __syncthreads();
  if (tid == 0) {
    __threadfence();
    unsigned old = atomicAdd(bar, 1u);
    unsigned target = (old / NWG) * NWG + NWG;
    while (__hip_atomic_load(bar, __ATOMIC_RELAXED, __HIP_MEMORY_SCOPE_AGENT) < target)
      __builtin_amdgcn_s_sleep(1);
  }
  __syncthreads();
  __threadfence();
}

__global__ void k_cvt_bf16(const float* __restrict__ s, unsigned short* __restrict__ d, int n) {
  for (int i = blockIdx.x * blockDim.x + threadIdx.x; i < n; i += gridDim.x * blockDim.x)
    d[i] = f2bf(s[i]);
}
__global__ void k_zero_f32(float* p, int n) {
  for (int i = blockIdx.x * blockDim.x + threadIdx.x; i < n; i += gridDim.x * blockDim.x)
    p[i] = 0.0f;
}
__global__ void k_zero_u16(unsigned short* p, int n) {
  for (int i = blockIdx.x * blockDim.x + threadIdx.x; i < n; i += gridDim.x * blockDim.x)
    p[i] = 0;
}

__global__ __launch_bounds__(TPB) void k_lstm_seq(
    const unsigned short* __restrict__ xbf,    // [B][T][D] bf16
    const unsigned short* __restrict__ Wih1,   // [4H][D]  bf16 (stays in L2)
    const unsigned short* __restrict__ Whh1,   // [4H][H]  bf16
    const unsigned short* __restrict__ Wih2,   // [4H][H]  bf16
    const unsigned short* __restrict__ Whh2,   // [4H][H]  bf16
    const float* __restrict__ bih1, const float* __restrict__ bhh1,
    const float* __restrict__ bih2, const float* __restrict__ bhh2,
    unsigned short* __restrict__ h1bf,         // [2][B][H] ping-pong bf16
    unsigned short* __restrict__ h2bf,         // [2][B][H] ping-pong bf16
    float* __restrict__ c1, float* __restrict__ c2,   // [B][H] f32
    unsigned short* __restrict__ h2seq,        // [T][B][H] bf16
    unsigned* bar)
{
  // Dynamic LDS layout (byte offsets; dynamic base assumed 0 — no static LDS):
  //   0      : Whh1 slice [4 gates][16 rows][512]   64KB
  //   65536  : Wih2 slice                           64KB
  //   131072 : Whh2 slice                           64KB
  //   196608 : actA [64][512]  (h1_prev / h1_new)   64KB
  //   262144 : actB [64][512]  (x tile / h2_prev)   64KB
  extern __shared__ unsigned short sm[];
  unsigned short* shWhh1 = sm;
  unsigned short* shWih2 = sm + 32768;
  unsigned short* shWhh2 = sm + 65536;
  unsigned short* actA   = sm + 98304;
  unsigned short* actB   = sm + 131072;

  const int tid   = threadIdx.x;
  const int lane  = tid & 31;
  const int mt    = tid >> 5;                 // wave = batch tile 0..3
  const int hct   = blockIdx.x;               // hidden-col tile 0..31
  const int arow  = lane & 15;
  const int mhalf = ((lane >> 4) & 1) * 8;    // C/D: lanes 16-31 -> rows +8
  const int ncol  = hct * 16 + arow;

  // ---- stage recurrent weight slices into LDS once (reused for all 512 t) ---
#ifdef HAVE_TDM
  if (tid < 32) {
    // 4 gate chunks of 16 contiguous rows (16KB), chunk stride = 512 rows.
    tdm_load_2d(0u,      Whh1 + (size_t)hct * 16 * NH, 16 * NH, 4, (size_t)512 * NH);
    tdm_load_2d(65536u,  Wih2 + (size_t)hct * 16 * NH, 16 * NH, 4, (size_t)512 * NH);
    tdm_load_2d(131072u, Whh2 + (size_t)hct * 16 * NH, 16 * NH, 4, (size_t)512 * NH);
    __builtin_amdgcn_s_wait_tensorcnt(0);
  }
#else
  for (int i = tid; i < 4096; i += TPB) {     // 4 chunks x 1024 uint4 per matrix
    const int g = i >> 10, q = i & 1023;
    const size_t go = ((size_t)(g * 512 + hct * 16)) * NH;
    ((uint4*)shWhh1)[i] = ((const uint4*)(Whh1 + go))[q];
    ((uint4*)shWih2)[i] = ((const uint4*)(Wih2 + go))[q];
    ((uint4*)shWhh2)[i] = ((const uint4*)(Whh2 + go))[q];
  }
#endif
  __syncthreads();

  for (int t = 0; t < NT; ++t) {
    const int rp = t & 1, wp = rp ^ 1;

    // ----------------------------- layer 1 -----------------------------
#ifdef HAVE_TDM
    if (tid < 32) {
      tdm_load_2d(196608u, h1bf + (size_t)rp * NB * NH, NB * NH, 1, NB * NH); // 64KB
      tdm_load_2d(262144u, xbf + (size_t)t * ND, ND, NB, (size_t)NT * ND);    // x tile
      __builtin_amdgcn_s_wait_tensorcnt(0);
    }
#else
    {
      const uint4* g = (const uint4*)(h1bf + (size_t)rp * NB * NH);
      uint4* s = (uint4*)actA;
      for (int i = tid; i < 4096; i += TPB) s[i] = g[i];
      for (int i = tid; i < 1024; i += TPB) {          // 64 rows x 16 uint4
        const int r = i >> 4, q = i & 15;
        ((uint4*)(actB + r * ND))[q] = ((const uint4*)(xbf + ((size_t)r * NT + t) * ND))[q];
      }
    }
#endif
    __syncthreads();

    v8f acc[4] = {};
    #pragma unroll
    for (int k0 = 0; k0 < ND; k0 += 32) {              // x @ Wih1^T, K=128
      v16bf a = ldfrag(actB + (mt * 16 + arow) * ND + k0, lane);
      #pragma unroll
      for (int g = 0; g < 4; ++g) {
        v16bf bm = ldfrag(Wih1 + ((size_t)(g * NH + ncol)) * ND + k0, lane);
        acc[g] = WMMA_BF16(a, bm, acc[g]);
      }
    }
    #pragma unroll 4
    for (int k0 = 0; k0 < NH; k0 += 32) {              // h1 @ Whh1^T, K=512
      v16bf a = ldfrag(actA + (mt * 16 + arow) * NH + k0, lane);
      #pragma unroll
      for (int g = 0; g < 4; ++g) {
        v16bf bm = ldfrag(shWhh1 + (g * 16 + arow) * NH + k0, lane);
        acc[g] = WMMA_BF16(a, bm, acc[g]);
      }
    }
    {
      float bsv[4];
      #pragma unroll
      for (int g = 0; g < 4; ++g) bsv[g] = bih1[g * NH + ncol] + bhh1[g * NH + ncol];
      #pragma unroll
      for (int j = 0; j < 8; ++j) {
        const int brow = mt * 16 + mhalf + j;
        const size_t ci = (size_t)brow * NH + ncol;
        const float iv = sigm (acc[0][j] + bsv[0]);
        const float fv = sigm (acc[1][j] + bsv[1]);
        const float gv = tanhf(acc[2][j] + bsv[2]);
        const float ov = sigm (acc[3][j] + bsv[3]);
        const float cn = fv * c1[ci] + iv * gv;
        c1[ci] = cn;
        h1bf[(size_t)wp * (NB * NH) + ci] = f2bf(ov * tanhf(cn));
      }
    }
    if (tid == 0 && t + 1 < NT)                        // global_prefetch_b8
      __builtin_prefetch(xbf + (size_t)(t + 1) * ND, 0, 1);
    gridbar(bar, tid);

    // ----------------------------- layer 2 -----------------------------
#ifdef HAVE_TDM
    if (tid < 32) {
      tdm_load_2d(196608u, h1bf + (size_t)wp * NB * NH, NB * NH, 1, NB * NH);
      tdm_load_2d(262144u, h2bf + (size_t)rp * NB * NH, NB * NH, 1, NB * NH);
      __builtin_amdgcn_s_wait_tensorcnt(0);
    }
#else
    {
      const uint4* g1p = (const uint4*)(h1bf + (size_t)wp * NB * NH);
      const uint4* g2p = (const uint4*)(h2bf + (size_t)rp * NB * NH);
      uint4* s1 = (uint4*)actA; uint4* s2 = (uint4*)actB;
      for (int i = tid; i < 4096; i += TPB) { s1[i] = g1p[i]; s2[i] = g2p[i]; }
    }
#endif
    __syncthreads();

    { v8f z = {}; acc[0] = z; acc[1] = z; acc[2] = z; acc[3] = z; }
    #pragma unroll 2
    for (int k0 = 0; k0 < NH; k0 += 32) {              // h1n@Wih2^T + h2@Whh2^T
      v16bf a1 = ldfrag(actA + (mt * 16 + arow) * NH + k0, lane);
      v16bf a2 = ldfrag(actB + (mt * 16 + arow) * NH + k0, lane);
      #pragma unroll
      for (int g = 0; g < 4; ++g) {
        v16bf b1 = ldfrag(shWih2 + (g * 16 + arow) * NH + k0, lane);
        acc[g] = WMMA_BF16(a1, b1, acc[g]);
        v16bf b2 = ldfrag(shWhh2 + (g * 16 + arow) * NH + k0, lane);
        acc[g] = WMMA_BF16(a2, b2, acc[g]);
      }
    }
    {
      float bsv[4];
      #pragma unroll
      for (int g = 0; g < 4; ++g) bsv[g] = bih2[g * NH + ncol] + bhh2[g * NH + ncol];
      #pragma unroll
      for (int j = 0; j < 8; ++j) {
        const int brow = mt * 16 + mhalf + j;
        const size_t ci = (size_t)brow * NH + ncol;
        const float iv = sigm (acc[0][j] + bsv[0]);
        const float fv = sigm (acc[1][j] + bsv[1]);
        const float gv = tanhf(acc[2][j] + bsv[2]);
        const float ov = sigm (acc[3][j] + bsv[3]);
        const float cn = fv * c2[ci] + iv * gv;
        c2[ci] = cn;
        const unsigned short hb = f2bf(ov * tanhf(cn));
        h2bf[(size_t)wp * (NB * NH) + ci] = hb;
        h2seq[((size_t)t * NB + brow) * NH + ncol] = hb;
      }
    }
    gridbar(bar, tid);
  }
}

// out[b][t][n] = h2seq[t][b][:] @ Wout[n][:] + bout[n];  M=T*B, N=1024, K=512
__global__ __launch_bounds__(256) void k_proj(
    const unsigned short* __restrict__ h2seq,
    const unsigned short* __restrict__ Wout,
    const float* __restrict__ bout,
    float* __restrict__ out)
{
  const int lane = threadIdx.x & 31;
  const int wid  = blockIdx.x * 8 + (threadIdx.x >> 5);
  const int mt   = wid >> 6;                 // 2048 m-tiles
  const int nt   = wid & 63;                 // 64 n-tiles
  const int arow = lane & 15;
  const unsigned short* abase = h2seq + ((size_t)(mt * 16 + arow)) * NH;
  const unsigned short* bbase = Wout  + ((size_t)(nt * 16 + arow)) * NH;
  v8f acc = {};
  #pragma unroll 4
  for (int k0 = 0; k0 < NH; k0 += 32) {
    v16bf a = ldfrag(abase + k0, lane);
    v16bf b = ldfrag(bbase + k0, lane);
    acc = WMMA_BF16(a, b, acc);
  }
  const int n = nt * 16 + arow;
  const float bb = bout[n];
  const int mhalf = ((lane >> 4) & 1) * 8;
  #pragma unroll
  for (int j = 0; j < 8; ++j) {
    const int m = mt * 16 + mhalf + j;       // m = t*B + b
    const int t = m >> 6, b = m & 63;
    out[((size_t)b * NT + t) * NO + n] = acc[j] + bb;
  }
}

extern "C" void kernel_launch(void* const* d_in, const int* in_sizes, int n_in,
                              void* d_out, int out_size, void* d_ws, size_t ws_size,
                              hipStream_t stream) {
  (void)in_sizes; (void)n_in; (void)out_size; (void)ws_size;
  const float* x    = (const float*)d_in[0];
  const float* Wih1 = (const float*)d_in[1];
  const float* Whh1 = (const float*)d_in[2];
  const float* bih1 = (const float*)d_in[3];
  const float* bhh1 = (const float*)d_in[4];
  const float* Wih2 = (const float*)d_in[5];
  const float* Whh2 = (const float*)d_in[6];
  const float* bih2 = (const float*)d_in[7];
  const float* bhh2 = (const float*)d_in[8];
  const float* Wout = (const float*)d_in[9];
  const float* bout = (const float*)d_in[10];
  float* out = (float*)d_out;

  char* ws = (char*)d_ws;
  size_t off = 0;
  auto take = [&](size_t bytes) {
    char* p = ws + off;
    off = (off + bytes + 255) & ~(size_t)255;
    return p;
  };
  unsigned short* xbf   = (unsigned short*)take((size_t)NB * NT * ND * 2);
  unsigned short* Wih1b = (unsigned short*)take((size_t)NG * ND * 2);
  unsigned short* Whh1b = (unsigned short*)take((size_t)NG * NH * 2);
  unsigned short* Wih2b = (unsigned short*)take((size_t)NG * NH * 2);
  unsigned short* Whh2b = (unsigned short*)take((size_t)NG * NH * 2);
  unsigned short* Woutb = (unsigned short*)take((size_t)NO * NH * 2);
  unsigned short* h2seq = (unsigned short*)take((size_t)NT * NB * NH * 2);
  unsigned short* h1bf  = (unsigned short*)take((size_t)2 * NB * NH * 2);
  unsigned short* h2bf  = (unsigned short*)take((size_t)2 * NB * NH * 2);
  float*    c1  = (float*)take((size_t)NB * NH * 4);
  float*    c2  = (float*)take((size_t)NB * NH * 4);
  unsigned* bar = (unsigned*)take(256);   // monotonic; start value must be a multiple of NWG

  auto cvt = [&](const float* s, unsigned short* d, int n) {
    int blocks = (n + 256 * 4 - 1) / (256 * 4);
    if (blocks > 4096) blocks = 4096;
    hipLaunchKernelGGL(k_cvt_bf16, dim3(blocks), dim3(256), 0, stream, s, d, n);
  };
  cvt(x,    xbf,   NB * NT * ND);
  cvt(Wih1, Wih1b, NG * ND);
  cvt(Whh1, Whh1b, NG * NH);
  cvt(Wih2, Wih2b, NG * NH);
  cvt(Whh2, Whh2b, NG * NH);
  cvt(Wout, Woutb, NO * NH);
  hipLaunchKernelGGL(k_zero_u16, dim3(64), dim3(256), 0, stream, h1bf, 2 * NB * NH);
  hipLaunchKernelGGL(k_zero_u16, dim3(64), dim3(256), 0, stream, h2bf, 2 * NB * NH);
  hipLaunchKernelGGL(k_zero_f32, dim3(64), dim3(256), 0, stream, c1, NB * NH);
  hipLaunchKernelGGL(k_zero_f32, dim3(64), dim3(256), 0, stream, c2, NB * NH);

  (void)hipFuncSetAttribute((const void*)k_lstm_seq,
                            hipFuncAttributeMaxDynamicSharedMemorySize,
                            (int)LDS_BYTES);
  hipLaunchKernelGGL(k_lstm_seq, dim3(NWG), dim3(TPB), LDS_BYTES, stream,
                     xbf, Wih1b, Whh1b, Wih2b, Whh2b,
                     bih1, bhh1, bih2, bhh2,
                     h1bf, h2bf, c1, c2, h2seq, bar);

  const int tiles = (NT * NB / 16) * (NO / 16);   // 131072 16x16 tiles
  hipLaunchKernelGGL(k_proj, dim3(tiles / 8), dim3(256), 0, stream,
                     h2seq, Woutb, bout, out);
}
